// EncoderInput_60979945668639
// MI455X (gfx1250) — matrix operand order repaired and checked by reading, not compiled, not run on hardware
//
#include <hip/hip_runtime.h>
#include <hip/hip_bf16.h>

// Problem dims (fixed by reference)
#define B_  64
#define T_  128
#define N_  128
#define M_  512
#define G4  (4 * M_)   // 2048 (gate dim)
#define M2  (2 * M_)   // 1024 (hc dim)

typedef __attribute__((ext_vector_type(16))) __bf16 v16bf;
typedef __attribute__((ext_vector_type(8)))  float  v8f;

__device__ __forceinline__ __bf16 f2bf(float f) {
  unsigned u = __builtin_bit_cast(unsigned, f);
  u += 0x7fffu + ((u >> 16) & 1u);              // round-to-nearest-even
  unsigned short s = (unsigned short)(u >> 16);
  return __builtin_bit_cast(__bf16, s);
}

__device__ __forceinline__ float fast_sigmoid(float x) {
  return 1.0f / (1.0f + __expf(-x));
}
__device__ __forceinline__ float fast_tanh(float x) {
  return 2.0f / (1.0f + __expf(-2.0f * x)) - 1.0f;
}

__device__ __forceinline__ v8f wmma_bf16(v16bf a, v16bf b, v8f c) {
  return __builtin_amdgcn_wmma_f32_16x16x32_bf16(
      /*neg_a=*/false, a, /*neg_b=*/false, b,
      /*c_mod=*/(short)0, c, /*reuse_a=*/false, /*reuse_b=*/false);
}

// Packed fragment: 512 bf16 per fragment (32 lanes x 16 elems), 32B per lane.
// Loading one fragment = two global_load_b128 per lane, zero conversion VALU.
__device__ __forceinline__ v16bf load_frag(const __bf16* __restrict__ p,
                                           int fragIdx) {
  int lane = threadIdx.x & 31;
  return *(const v16bf*)(p + ((size_t)fragIdx << 9) + (lane << 4));
}

// ---- Packing kernels (one pass; everything stays hot in 192MB L2) ----------
// B-matrix pack: src f32 (K x C row-major) -> WMMA B-fragment layout.
// frag index = kt * (C/16) + ct ; lane n=lane&15, g=lane>>4 ; K(e)=kt*32+16*(e>>3)+8g+(e&7)
__global__ void __launch_bounds__(256)
pack_b_kernel(const float* __restrict__ src, __bf16* __restrict__ dst,
              int K, int C) {
  int i = blockIdx.x * blockDim.x + threadIdx.x;
  if (i >= K * C) return;
  int frag = i >> 9;
  int lane = (i >> 4) & 31;
  int e    = i & 15;
  int n = lane & 15, g = lane >> 4;
  int ctiles = C >> 4;
  int kt = frag / ctiles, ct = frag % ctiles;
  int k = kt * 32 + ((e >> 3) << 4) + (g << 3) + (e & 7);
  dst[i] = f2bf(src[(size_t)k * C + ct * 16 + n]);
}

// A-matrix pack: src f32 with arbitrary (rstride,kstride) and optional batch.
// frag index = rt * (K/32) + kt ; lane m=lane&15, g ; K(e)=kt*32+16*(e>>3)+8g+(e&7)
__global__ void __launch_bounds__(256)
pack_a_kernel(const float* __restrict__ src, __bf16* __restrict__ dst,
              int R, int K, int rstride, int kstride,
              int nbatch, long srcBatchStride, long dstBatchStride) {
  long i = (long)blockIdx.x * blockDim.x + threadIdx.x;
  long per = (long)R * K;
  if (i >= per * nbatch) return;
  int  bidx = (int)(i / per);
  long j    = i % per;
  int frag = (int)(j >> 9);
  int lane = ((int)j >> 4) & 31;
  int e    = (int)j & 15;
  int m = lane & 15, g = lane >> 4;
  int ktiles = K >> 5;
  int rt = frag / ktiles, kt = frag % ktiles;
  int k = kt * 32 + ((e >> 3) << 4) + (g << 3) + (e & 7);
  dst[bidx * dstBatchStride + j] =
      f2bf(src[bidx * srcBatchStride + (size_t)(rt * 16 + m) * rstride +
               (size_t)k * kstride]);
}

// Scatter one value into packed A-fragment layout (used by gates epilogue so
// the hc @ W_e GEMM reads pure b128 fragments).  ktiles for hc = 1024/32 = 32.
__device__ __forceinline__ void store_packed_hc(__bf16* __restrict__ pHC,
                                                int row, int col, float v) {
  int rt = row >> 4, m = row & 15;
  int kt = col >> 5, kk = col & 31;
  int g = (kk >> 3) & 1;
  int e = ((kk >> 4) << 3) | (kk & 7);
  int lane = m + (g << 4);
  pHC[(((size_t)rt * 32 + kt) << 9) + (lane << 4) + e] = f2bf(v);
}

// ---- K0: hb = h0 @ U_lstm + b_lstm   (64 x 2048, K=512) --------------------
__global__ void __launch_bounds__(128)
hb_kernel(const __bf16* __restrict__ pH0, const __bf16* __restrict__ pUl,
          const float* __restrict__ b_lstm, float* __restrict__ hb) {
  int wid = blockIdx.x * (blockDim.x >> 5) + (threadIdx.x >> 5);
  const int ntiles = G4 / 16;          // 128
  int tm = wid / ntiles, tn = wid % ntiles;
  v8f acc = {};
#pragma unroll 4
  for (int kt = 0; kt < 16; ++kt) {    // K=512 -> 16 chunks of 32
    v16bf a = load_frag(pH0, tm * 16 + kt);
    v16bf b = load_frag(pUl, kt * 128 + tn);
    acc = wmma_bf16(a, b, acc);
  }
  int lane = threadIdx.x & 31;
  int n = lane & 15, g = lane >> 4;
  float bias = b_lstm[tn * 16 + n];
#pragma unroll
  for (int r = 0; r < 8; ++r) {
    int row = tm * 16 + 8 * g + r;
    hb[(size_t)row * G4 + tn * 16 + n] = acc[r] + bias;
  }
}

// ---- K1: z = X @ W_lstm + hb; gates; write packed hc -----------------------
__global__ void __launch_bounds__(128)
gates_kernel(const __bf16* __restrict__ pXA, const __bf16* __restrict__ pWl,
             const float* __restrict__ hb, const float* __restrict__ s0,
             __bf16* __restrict__ pHC) {
  int wid = blockIdx.x * (blockDim.x >> 5) + (threadIdx.x >> 5);
  const int mtiles = M_ / 16;          // 32
  int tr = wid / mtiles, tc = wid % mtiles;
  int row0 = tr * 16;                  // row in [0, B*T)
  int mcol0 = tc * 16;
  int b = row0 / T_;                   // 16-row tile lies in one batch
  v8f accI = {}, accF = {}, accC = {}, accO = {};
#pragma unroll
  for (int kt = 0; kt < 4; ++kt) {     // K=128 -> 4 chunks of 32
    v16bf a  = load_frag(pXA, tr * 4 + kt);
    if (kt < 3)
      __builtin_prefetch(pXA + (((size_t)(tr * 4 + kt + 1)) << 9) +
                             ((threadIdx.x & 31) << 4), 0, 0);
    v16bf bI = load_frag(pWl, kt * 128 + 0 * 32 + tc);
    v16bf bF = load_frag(pWl, kt * 128 + 1 * 32 + tc);
    v16bf bC = load_frag(pWl, kt * 128 + 2 * 32 + tc);
    v16bf bO = load_frag(pWl, kt * 128 + 3 * 32 + tc);
    accI = wmma_bf16(a, bI, accI);
    accF = wmma_bf16(a, bF, accF);
    accC = wmma_bf16(a, bC, accC);
    accO = wmma_bf16(a, bO, accO);
  }
  int lane = threadIdx.x & 31;
  int n = lane & 15, g = lane >> 4;
  int m = mcol0 + n;
  float hbI = hb[(size_t)b * G4 + 0 * M_ + m];
  float hbF = hb[(size_t)b * G4 + 1 * M_ + m];
  float hbC = hb[(size_t)b * G4 + 2 * M_ + m];
  float hbO = hb[(size_t)b * G4 + 3 * M_ + m];
  float s0v = s0[(size_t)b * M_ + m];
#pragma unroll
  for (int r = 0; r < 8; ++r) {
    float zi = accI[r] + hbI, zf = accF[r] + hbF;
    float zc = accC[r] + hbC, zo = accO[r] + hbO;
    float cs = fast_sigmoid(zf) * s0v + fast_sigmoid(zi) * fast_tanh(zc);
    float hv = fast_sigmoid(zo) * fast_tanh(cs);
    int row = row0 + 8 * g + r;
    store_packed_hc(pHC, row, m, hv);        // h  -> cols [0, 512)
    store_packed_hc(pHC, row, M_ + m, cs);   // c  -> cols [512, 1024)
  }
}

// ---- K2: A = hc @ W_e + b_We   (8192 x 128, K=1024) ------------------------
__global__ void __launch_bounds__(128)
attA_kernel(const __bf16* __restrict__ pHC, const __bf16* __restrict__ pWe,
            const float* __restrict__ b_We, float* __restrict__ Amat) {
  int wid = blockIdx.x * (blockDim.x >> 5) + (threadIdx.x >> 5);
  const int ntiles = T_ / 16;          // 8
  int tr = wid / ntiles, tn = wid % ntiles;
  v8f acc = {};
#pragma unroll 4
  for (int kt = 0; kt < 32; ++kt) {    // K=1024 -> 32 chunks of 32
    v16bf a = load_frag(pHC, tr * 32 + kt);
    if (kt < 31)
      __builtin_prefetch(pHC + (((size_t)(tr * 32 + kt + 1)) << 9) +
                             ((threadIdx.x & 31) << 4), 0, 0);
    v16bf b = load_frag(pWe, kt * 8 + tn);
    acc = wmma_bf16(a, b, acc);
  }
  int lane = threadIdx.x & 31;
  int n = lane & 15, g = lane >> 4;
  float bias = b_We[tn * 16 + n];
#pragma unroll
  for (int r = 0; r < 8; ++r) {
    int row = tr * 16 + 8 * g + r;
    Amat[(size_t)row * T_ + tn * 16 + n] = acc[r] + bias;
  }
}

// ---- K3: U[b] = X[b]^T @ U_e + b_Ue  (per b: 128x128, K=128) ---------------
__global__ void __launch_bounds__(128)
attU_kernel(const __bf16* __restrict__ pXT, const __bf16* __restrict__ pUe,
            const float* __restrict__ b_Ue, float* __restrict__ Uu) {
  int wid = blockIdx.x * (blockDim.x >> 5) + (threadIdx.x >> 5);
  const int nt = N_ / 16, tt = T_ / 16;         // 8, 8
  int bidx = wid / (nt * tt);
  int rem  = wid % (nt * tt);
  int trn = rem / tt, tck = rem % tt;
  const __bf16* pA = pXT + (size_t)bidx * (N_ * T_);
  v8f acc = {};
#pragma unroll
  for (int kt = 0; kt < 4; ++kt) {     // K=128 -> 4 chunks of 32
    v16bf a = load_frag(pA, trn * 4 + kt);
    v16bf b = load_frag(pUe, kt * 8 + tck);
    acc = wmma_bf16(a, b, acc);
  }
  int lane = threadIdx.x & 31;
  int n = lane & 15, g = lane >> 4;
  float bias = b_Ue[tck * 16 + n];
#pragma unroll
  for (int r = 0; r < 8; ++r) {
    int row = trn * 16 + 8 * g + r;              // n-dimension index
    Uu[((size_t)bidx * N_ + row) * T_ + tck * 16 + n] = acc[r] + bias;
  }
}

// ---- K4: fused scores + softmax + scale (never materialize B,T,N,T) --------
__global__ void __launch_bounds__(128)
score_kernel(const float* __restrict__ X, const float* __restrict__ Amat,
             const float* __restrict__ Uu, const float* __restrict__ v_e,
             const float* __restrict__ b_ve, float* __restrict__ out) {
  __shared__ float sA[T_];
  __shared__ float sV[T_];
  __shared__ float sR[N_];
  int bt = blockIdx.x;                 // 0 .. B*T-1
  int b  = bt / T_;
  int n  = threadIdx.x;                // 0 .. 127
  sA[n] = Amat[(size_t)bt * T_ + n];
  sV[n] = v_e[n];
  __syncthreads();
  const float* Urow = Uu + ((size_t)b * N_ + n) * T_;
  float acc = 0.f;
#pragma unroll 4
  for (int k = 0; k < T_; ++k)
    acc += fast_tanh(sA[k] + Urow[k]) * sV[k];
  float score = acc + b_ve[0];
  sR[n] = score;
  __syncthreads();
  for (int s = 64; s > 0; s >>= 1) {
    if (n < s) sR[n] = fmaxf(sR[n], sR[n + s]);
    __syncthreads();
  }
  float mx = sR[0];
  __syncthreads();
  float ex = __expf(score - mx);
  sR[n] = ex;
  __syncthreads();
  for (int s = 64; s > 0; s >>= 1) {
    if (n < s) sR[n] += sR[n + s];
    __syncthreads();
  }
  float inv = 1.0f / sR[0];
  out[(size_t)bt * N_ + n] = X[(size_t)bt * N_ + n] * ex * inv;
}

// ---------------------------------------------------------------------------
extern "C" void kernel_launch(void* const* d_in, const int* in_sizes, int n_in,
                              void* d_out, int out_size, void* d_ws, size_t ws_size,
                              hipStream_t stream) {
  const float* X      = (const float*)d_in[0];
  const float* h0     = (const float*)d_in[1];
  const float* s0     = (const float*)d_in[2];
  const float* W_lstm = (const float*)d_in[3];
  const float* U_lstm = (const float*)d_in[4];
  const float* b_lstm = (const float*)d_in[5];
  const float* W_e    = (const float*)d_in[6];
  const float* b_We   = (const float*)d_in[7];
  const float* U_e    = (const float*)d_in[8];
  const float* b_Ue   = (const float*)d_in[9];
  const float* v_e    = (const float*)d_in[10];
  const float* b_ve   = (const float*)d_in[11];
  float* out = (float*)d_out;

  // Workspace layout (~31.4 MB, all offsets 4KB-aligned for b128 frag loads):
  char* ws = (char*)d_ws;
  float*  hb   = (float*) (ws + 0x0000000);  // 64*2048 f32      512KB
  float*  Amat = (float*) (ws + 0x0080000);  // 8192*128 f32       4MB
  float*  Uu   = (float*) (ws + 0x0480000);  // 64*128*128 f32     4MB
  __bf16* pHC  = (__bf16*)(ws + 0x0880000);  // packed hc A-frag  16MB
  __bf16* pXA  = (__bf16*)(ws + 0x1880000);  // packed X  A-frag   2MB
  __bf16* pXT  = (__bf16*)(ws + 0x1A80000);  // packed X^T A-frag  2MB
  __bf16* pUl  = (__bf16*)(ws + 0x1C80000);  // packed U_lstm B    2MB
  __bf16* pWl  = (__bf16*)(ws + 0x1E80000);  // packed W_lstm B  512KB
  __bf16* pWe  = (__bf16*)(ws + 0x1F00000);  // packed W_e B     256KB
  __bf16* pUe  = (__bf16*)(ws + 0x1F40000);  // packed U_e B      32KB
  __bf16* pH0  = (__bf16*)(ws + 0x1F48000);  // packed h0 A       64KB

  // --- pack phase (single pass over each operand) ---
  hipLaunchKernelGGL(pack_b_kernel, dim3(1024), dim3(256), 0, stream,
                     W_lstm, pWl, N_, G4);                       // 128 x 2048
  hipLaunchKernelGGL(pack_b_kernel, dim3(4096), dim3(256), 0, stream,
                     U_lstm, pUl, M_, G4);                       // 512 x 2048
  hipLaunchKernelGGL(pack_b_kernel, dim3(512), dim3(256), 0, stream,
                     W_e, pWe, M2, T_);                          // 1024 x 128
  hipLaunchKernelGGL(pack_b_kernel, dim3(64), dim3(256), 0, stream,
                     U_e, pUe, T_, T_);                          // 128 x 128
  hipLaunchKernelGGL(pack_a_kernel, dim3(4096), dim3(256), 0, stream,
                     X, pXA, B_ * T_, N_, N_, 1, 1, 0L, 0L);     // X rows
  hipLaunchKernelGGL(pack_a_kernel, dim3(4096), dim3(256), 0, stream,
                     X, pXT, N_, T_, 1, N_, B_,
                     (long)T_ * N_, (long)N_ * T_);              // X^T per b
  hipLaunchKernelGGL(pack_a_kernel, dim3(128), dim3(256), 0, stream,
                     h0, pH0, B_, M_, M_, 1, 1, 0L, 0L);         // h0 rows

  // --- GEMM + fused phases ---
  hipLaunchKernelGGL(hb_kernel, dim3(128), dim3(128), 0, stream,
                     pH0, pUl, b_lstm, hb);                      // 512 waves
  hipLaunchKernelGGL(gates_kernel, dim3(4096), dim3(128), 0, stream,
                     pXA, pWl, hb, s0, pHC);                     // 16384 waves
  hipLaunchKernelGGL(attA_kernel, dim3(1024), dim3(128), 0, stream,
                     pHC, pWe, b_We, Amat);                      // 4096 waves
  hipLaunchKernelGGL(attU_kernel, dim3(1024), dim3(128), 0, stream,
                     pXT, pUe, b_Ue, Uu);                        // 4096 waves
  hipLaunchKernelGGL(score_kernel, dim3(B_ * T_), dim3(128), 0, stream,
                     X, Amat, Uu, v_e, b_ve, out);
}
